// DCN_81423989998096
// MI455X (gfx1250) — compile-verified, matrix operand order checked by reading, not run on hardware
//
#include <hip/hip_runtime.h>
#include <hip/hip_bf16.h>
#include <math.h>

typedef __attribute__((ext_vector_type(16))) _Float16 v16h;
typedef __attribute__((ext_vector_type(8)))  float    v8f;

#define CIN     256
#define COUT    256
#define KK      9
#define KGLOB   (CIN * KK)      // 2304
#define NKT     (KGLOB / 32)    // 72 k-chunks of 32; kk-major: kk = kt>>3, cbase = (kt&7)*32
#define HW      4096
#define BATCH   4
#define NT_PIX  32              // pixel (N) tile: two 16-wide WMMA N-subtiles
#define OM_ROWS 32              // 27 offset/mask channels padded to 32
#define KT_SEC  18              // K-chunks staged in LDS per section (4 sections)

// (lane, half-element) -> K offset within a 32-wide chunk, per ISA 16-bit A/B
// 16x32 layout: lanes 0-15 hold K {0..7,16..23}, lanes 16-31 hold {8..15,24..31}
__device__ __forceinline__ int kmap(int lane, int e) {
    return ((e < 8) ? e : (e + 8)) + ((lane >= 16) ? 8 : 0);
}

// ---------------------------------------------------------------------------
// Kernel 1: repack f32 weights (OIHW) into WMMA A layout with kk-major K:
//   packed K index K' = kk*256 + c   (chunk kt: kk = kt>>3, c = (kt&7)*32+off)
//   original flat k  = c*9 + kk
// pw  [16 m-tiles][72 kt][32 lanes][16 halves]  (main conv, 256 x 2304)
// pow [ 2 m-tiles][72 kt][32 lanes][16 halves]  (offset conv, 27 x 2304, pad)
// ---------------------------------------------------------------------------
__global__ __launch_bounds__(256)
void pack_weights(const float* __restrict__ w,       // [256][2304]
                  const float* __restrict__ omw,     // [27][2304]
                  _Float16* __restrict__ pw,
                  _Float16* __restrict__ pow_) {
    int i = blockIdx.x * blockDim.x + threadIdx.x;
    const int NMAIN = 16 * NKT * 512;
    const int NOM   =  2 * NKT * 512;
    if (i < NMAIN) {
        int e = i & 15, lane = (i >> 4) & 31;
        int kt = (i >> 9) % NKT, mt = i / (512 * NKT);
        int kk = kt >> 3;
        int c  = (kt & 7) * 32 + kmap(lane, e);
        int o  = mt * 16 + (lane & 15);
        pw[i] = (_Float16)w[o * KGLOB + c * KK + kk];
    } else if (i < NMAIN + NOM) {
        int j = i - NMAIN;
        int e = j & 15, lane = (j >> 4) & 31;
        int kt = (j >> 9) % NKT, mt = j / (512 * NKT);
        int kk = kt >> 3;
        int c  = (kt & 7) * 32 + kmap(lane, e);
        int o  = mt * 16 + (lane & 15);
        pow_[j] = (o < 27) ? (_Float16)omw[o * KGLOB + c * KK + kk] : (_Float16)0.f;
    }
}

// ---------------------------------------------------------------------------
// Kernel 2: offset/mask conv as WMMA implicit GEMM.  M=32 (2 waves), N=16 px.
// kk constant per K-chunk -> branchless im2col, one v16h store per lane row.
// Output: om_out [B][32][4096] f32, bias added.
// ---------------------------------------------------------------------------
__global__ __launch_bounds__(64)
void om_gemm(const float* __restrict__ x,          // [4][256][64][64]
             const _Float16* __restrict__ pow_,    // packed om weights
             const float* __restrict__ om_bias,    // [27]
             float* __restrict__ om_out) {         // [4][32][4096]
    __shared__ __align__(32) _Float16 sPanel[512];
    int t   = blockIdx.x;
    int b   = t >> 8;
    int hw0 = (t & 255) << 4;          // 16 pixels, all on one image row
    int h   = hw0 >> 6, w0 = hw0 & 63;
    int tid = threadIdx.x, lane = tid & 31, wave = tid >> 5;   // wave 0..1
    v8f acc = {};
    for (int kt = 0; kt < NKT; ++kt) {
        int kk  = kt >> 3;
        int dy  = (kk / 3) - 1;
        int dxk = (kk % 3) - 1;
        // build 32(K) x 16(N) f16 B panel: 32 lane-vectors of 16 halves
        for (int vec = tid; vec < 32; vec += 64) {
            int N     = vec & 15;
            int cbase = (kt & 7) * 32 + ((vec >= 16) ? 8 : 0);
            int yy    = h + dy;
            int xx    = w0 + N + dxk;
            bool valid = (yy >= 0) && (yy < 64) && (xx >= 0) && (xx < 64);
            int idx = min(max(yy, 0), 63) * 64 + min(max(xx, 0), 63);
            const float* xb = x + (b * CIN + cbase) * HW + idx;
            v16h vv;
            #pragma unroll
            for (int e = 0; e < 16; ++e) {
                int coff = (e < 8) ? e : (e + 8);
                float v = xb[coff * HW];
                vv[e] = (_Float16)(valid ? v : 0.f);
            }
            *(v16h*)&sPanel[vec * 16] = vv;
        }
        __syncthreads();
        v16h bv = *(const v16h*)&sPanel[lane * 16];
        v16h av = *(const v16h*)&pow_[((wave * NKT + kt) * 32 + lane) * 16];
        acc = __builtin_amdgcn_wmma_f32_16x16x32_f16(false, av, false, bv,
                                                     (short)0, acc, false, false);
        __syncthreads();
    }
    int moff = (lane >= 16) ? 8 : 0;
    int n = lane & 15;
    #pragma unroll
    for (int r = 0; r < 8; ++r) {
        int M = wave * 16 + moff + r;
        if (M < 27)
            om_out[(b * OM_ROWS + M) * HW + hw0 + n] = acc[r] + om_bias[M];
    }
}

// ---------------------------------------------------------------------------
// Kernel 3: fused deformable im2col + main GEMM.
// Block = 256 threads (8 waves), one 32-pixel tile, all 256 output channels.
// Phase 0: 288 (tap,pixel) bilinear index/weight sets (mask folded in).
// Phase 1: build f16 cols lane-vectors in LDS (WMMA B layout), two 16-pixel
//          N-subtiles per K-chunk; one param fetch per 16 chans (two runs of 8).
// Phase 2: each wave owns 2 M-tiles x 2 N-subtiles -> 4 WMMAs per K-chunk
//          while fetching A only once per M-tile (2:1 better intensity).
// ---------------------------------------------------------------------------
__global__ __launch_bounds__(256)
void dcn_main(const float* __restrict__ x,         // [4][256][64][64]
              const _Float16* __restrict__ pw,     // packed main weights
              const float* __restrict__ bias,      // [256]
              const float* __restrict__ om,        // [4][32][4096]
              float* __restrict__ out) {           // [4][256][64][64]
    __shared__ __align__(32) _Float16 sCols[KT_SEC * 2 * 512];
    __shared__ __align__(16) int   sIdx[KK * NT_PIX * 4];
    __shared__ __align__(16) float sWt [KK * NT_PIX * 4];
    int t   = blockIdx.x;
    int b   = t >> 7;
    int hw0 = (t & 127) << 5;          // 32 pixels, all on one image row
    int h   = hw0 >> 6, w0 = hw0 & 63;
    int tid = threadIdx.x, lane = tid & 31, wave = tid >> 5;   // wave 0..7

    // ---- phase 0: sampling parameters, one thread per (tap, pixel) ----
    for (int p = tid; p < KK * NT_PIX; p += 256) {
        int kk = p >> 5, pix = p & 31;
        int hw = hw0 + pix;
        float dy = om[(b * OM_ROWS + 2 * kk    ) * HW + hw];
        float dx = om[(b * OM_ROWS + 2 * kk + 1) * HW + hw];
        float mr = om[(b * OM_ROWS + 18 + kk   ) * HW + hw];
        float mask = 1.f / (1.f + __expf(-mr));
        float py = dy + (float)(kk / 3) + (float)(h - 1);
        float px = dx + (float)(kk % 3) + (float)(w0 + pix - 1);
        float y0f = floorf(py), x0f = floorf(px);
        float fy = py - y0f, fx = px - x0f;
        int y0 = (int)y0f, x0 = (int)x0f;
        int base = p * 4;
        #pragma unroll
        for (int cn = 0; cn < 4; ++cn) {
            int yy = y0 + (cn >> 1);
            int xx = x0 + (cn & 1);
            float wy = (cn >> 1) ? fy : (1.f - fy);
            float wx = (cn & 1)  ? fx : (1.f - fx);
            bool valid = (yy >= 0) && (yy < 64) && (xx >= 0) && (xx < 64);
            int yc = min(max(yy, 0), 63), xc = min(max(xx, 0), 63);
            sIdx[base + cn] = yc * 64 + xc;
            sWt [base + cn] = valid ? (wy * wx * mask) : 0.f;
        }
    }
    __syncthreads();

    v8f acc00 = {}, acc01 = {}, acc10 = {}, acc11 = {};
    int mt0 = wave * 2, mt1 = wave * 2 + 1;

    for (int sect = 0; sect < 4; ++sect) {
        int ktBase = sect * KT_SEC;
        // ---- phase 1: one thread per 16-half lane vector ----
        for (int vec = tid; vec < KT_SEC * 64; vec += 256) {
            int ls  = vec & 31;
            int ns  = (vec >> 5) & 1;      // N-subtile
            int ktl = vec >> 6;
            int kt  = ktBase + ktl;
            int kk  = kt >> 3;
            int N   = ns * 16 + (ls & 15);
            int cbase = (kt & 7) * 32 + ((ls >= 16) ? 8 : 0);
            int4   ii = *(const int4*)  &sIdx[(kk * NT_PIX + N) * 4];
            float4 ww = *(const float4*)&sWt [(kk * NT_PIX + N) * 4];
            const float* xb = x + (b * CIN + cbase) * HW;
            v16h vv;
            #pragma unroll
            for (int e = 0; e < 16; ++e) {
                int coff = (e < 8) ? e : (e + 8);
                const float* xc = xb + coff * HW;
                float v = ww.x * xc[ii.x] + ww.y * xc[ii.y]
                        + ww.z * xc[ii.z] + ww.w * xc[ii.w];
                vv[e] = (_Float16)v;
            }
            *(v16h*)&sCols[((ktl * 2 + ns) * 32 + ls) * 16] = vv;
        }
        __syncthreads();
        // ---- phase 2: GEMM over this section of K ----
        for (int ktl = 0; ktl < KT_SEC; ++ktl) {
            int kt = ktBase + ktl;
            v16h b0 = *(const v16h*)&sCols[(ktl * 2 + 0) * 512 + lane * 16];
            v16h b1 = *(const v16h*)&sCols[(ktl * 2 + 1) * 512 + lane * 16];
            v16h a0 = *(const v16h*)&pw[((mt0 * NKT + kt) * 32 + lane) * 16];
            v16h a1 = *(const v16h*)&pw[((mt1 * NKT + kt) * 32 + lane) * 16];
            acc00 = __builtin_amdgcn_wmma_f32_16x16x32_f16(false, a0, false, b0,
                                                           (short)0, acc00, false, false);
            acc01 = __builtin_amdgcn_wmma_f32_16x16x32_f16(false, a0, false, b1,
                                                           (short)0, acc01, false, false);
            acc10 = __builtin_amdgcn_wmma_f32_16x16x32_f16(false, a1, false, b0,
                                                           (short)0, acc10, false, false);
            acc11 = __builtin_amdgcn_wmma_f32_16x16x32_f16(false, a1, false, b1,
                                                           (short)0, acc11, false, false);
        }
        __syncthreads();
    }

    // ---- phase 3: store D + bias (C/D layout: M = mt*16 + r + 8*(lane>=16)) ----
    int n = lane & 15;
    int moff = (lane >= 16) ? 8 : 0;
    #pragma unroll
    for (int r = 0; r < 8; ++r) {
        int M0 = mt0 * 16 + moff + r;
        int M1 = mt1 * 16 + moff + r;
        float b0v = bias[M0], b1v = bias[M1];
        out[(b * COUT + M0) * HW + hw0 + n]      = acc00[r] + b0v;
        out[(b * COUT + M0) * HW + hw0 + 16 + n] = acc01[r] + b0v;
        out[(b * COUT + M1) * HW + hw0 + n]      = acc10[r] + b1v;
        out[(b * COUT + M1) * HW + hw0 + 16 + n] = acc11[r] + b1v;
    }
}

// ---------------------------------------------------------------------------
extern "C" void kernel_launch(void* const* d_in, const int* in_sizes, int n_in,
                              void* d_out, int out_size, void* d_ws, size_t ws_size,
                              hipStream_t stream) {
    const float* x      = (const float*)d_in[0];   // [4,256,64,64]
    const float* weight = (const float*)d_in[1];   // [256,256,3,3]
    const float* bias   = (const float*)d_in[2];   // [256]
    const float* om_w   = (const float*)d_in[3];   // [27,256,3,3]
    const float* om_b   = (const float*)d_in[4];   // [27]
    float* out = (float*)d_out;                    // [4,256,64,64]

    // workspace layout (all offsets 256B-aligned):
    //   [0, 1179648)           packed main weight f16  (16*72*512 halves)
    //   [1179648, 1327104)     packed om weight f16    ( 2*72*512 halves)
    //   [1327104, 3424256)     om conv output f32      (4*32*4096 floats)
    char* ws = (char*)d_ws;
    _Float16* pw     = (_Float16*)ws;
    _Float16* pow_   = (_Float16*)(ws + 1179648);
    float*    om_out = (float*)   (ws + 1327104);

    int total = 16 * NKT * 512 + 2 * NKT * 512;
    pack_weights<<<(total + 255) / 256, 256, 0, stream>>>(weight, om_w, pw, pow_);
    om_gemm<<<BATCH * 256, 64, 0, stream>>>(x, pow_, om_b, om_out);
    dcn_main<<<BATCH * 128, 256, 0, stream>>>(x, pw, bias, om_out, out);
}
